// IPGN_21827023798629
// MI455X (gfx1250) — matrix operand order repaired and checked by reading, not compiled, not run on hardware
//
#include <hip/hip_runtime.h>
#include <hip/hip_bf16.h>

typedef __attribute__((ext_vector_type(16))) _Float16 v16h;
typedef __attribute__((ext_vector_type(8)))  _Float16 v8h;
typedef __attribute__((ext_vector_type(8)))  float    v8f;

#define NSAMPLE   32
#define RADIUS2   0.04f
#define S_NODES   519
#define N_POINTS  8192
#define DIM       128

// ---------------------------------------------------------------------------
// WMMA helpers (CDNA5, wave32, v_wmma_f32_16x16x32_f16)
// ---------------------------------------------------------------------------
static __device__ __forceinline__ v8f wmma16(v16h a, v16h b, v8f c) {
  return __builtin_amdgcn_wmma_f32_16x16x32_f16(false, a, false, b, (short)0, c,
                                                false, false);
}

// A-fragment (16x32 f16, ISA 7.12.2): lane<16 holds row (lane), K = {0..7,16..23};
// lane>=16 holds row (lane-16), K = {8..15,24..31}. Xs is row-major 32 x K in LDS.
static __device__ __forceinline__ v16h load_a_frag(const _Float16* Xs, int K,
                                                   int mt, int kt, int lane) {
  int m = mt * 16 + (lane & 15);
  const _Float16* p = Xs + m * K + kt * 32 + ((lane >> 4) << 3);
  v8h lo = *(const v8h*)p;          // K offsets hi*8 .. hi*8+7
  v8h hi = *(const v8h*)(p + 16);   // K offsets hi*8+16 .. hi*8+23
  return __builtin_shufflevector(lo, hi, 0,1,2,3,4,5,6,7,8,9,10,11,12,13,14,15);
}

// ---------------------------------------------------------------------------
// Weight -> B-fragment layout conversion (f32 KxN -> f16 fragments).
// B 32x16: lane l holds column l&15, K = (e>=8?16:0)+(l>=16?8:0)+(e&7).
// Fragment f = kt*(N/16)+nt stored as 32 lanes x 16 halfs contiguous.
// ---------------------------------------------------------------------------
__global__ void wfrag_kernel(const float* __restrict__ W, _Float16* __restrict__ F,
                             int K, int N) {
  int t = blockIdx.x * blockDim.x + threadIdx.x;
  if (t >= K * N) return;
  int e  = t & 15;
  int l  = (t >> 4) & 31;
  int f  = t >> 9;
  int NT = N >> 4;
  int nt = f % NT;
  int kt = f / NT;
  int k  = kt * 32 + ((e >> 3) << 4) + ((l >> 4) << 3) + (e & 7);
  int n  = nt * 16 + (l & 15);
  F[t] = (_Float16)W[(size_t)k * N + n];
}

// ---------------------------------------------------------------------------
// Ball query: first NSAMPLE in-range indices in ascending order, pad with first
// (pad with n-1 when none) — matches reference sort/where/clip semantics.
// ---------------------------------------------------------------------------
__global__ void query_ball_kernel(const float* __restrict__ data, int n,
                                  const float* __restrict__ query, int m,
                                  int* __restrict__ out) {
  int t = blockIdx.x * blockDim.x + threadIdx.x;
  if (t >= m) return;
  float qx = query[t * 3 + 0], qy = query[t * 3 + 1], qz = query[t * 3 + 2];
  int cnt = 0, first = n - 1;
  for (int i = 0; i < n && cnt < NSAMPLE; i++) {
    float dx = data[i * 3 + 0] - qx;
    float dy = data[i * 3 + 1] - qy;
    float dz = data[i * 3 + 2] - qz;
    float d2 = dx * dx + dy * dy + dz * dz;
    if (d2 <= RADIUS2) {
      if (cnt == 0) first = i;
      out[t * NSAMPLE + cnt] = i;
      cnt++;
    }
  }
  for (int j = cnt; j < NSAMPLE; j++) out[t * NSAMPLE + j] = first;
}

// ---------------------------------------------------------------------------
// One 32x128 @ 128x128 layer (relu, f16 out to LDS) — used by group MLP.
// 4 waves, wave w handles n-tiles w, w+4.
// ---------------------------------------------------------------------------
static __device__ __forceinline__ void mlp_layer128(const _Float16* Xs,
                                                    const _Float16* __restrict__ Wf,
                                                    const float* __restrict__ bias,
                                                    _Float16* Xd, int lane, int wv) {
  for (int nt = wv; nt < 8; nt += 4) {
    v8f c0 = {0.f,0.f,0.f,0.f,0.f,0.f,0.f,0.f};
    v8f c1 = {0.f,0.f,0.f,0.f,0.f,0.f,0.f,0.f};
    for (int kt = 0; kt < 4; kt++) {
      v16h a0 = load_a_frag(Xs, 128, 0, kt, lane);
      v16h a1 = load_a_frag(Xs, 128, 1, kt, lane);
      v16h b  = *(const v16h*)(Wf + (size_t)(kt * 8 + nt) * 512 + lane * 16);
      c0 = wmma16(a0, b, c0);
      c1 = wmma16(a1, b, c1);
    }
    int n = nt * 16 + (lane & 15);
    float bv = bias[n];
    int moff = (lane & 16) ? 8 : 0;
    for (int r = 0; r < 8; r++) {
      float v0 = fmaxf(c0[r] + bv, 0.f);
      float v1 = fmaxf(c1[r] + bv, 0.f);
      Xd[(r + moff) * 128 + n]      = (_Float16)v0;
      Xd[(16 + r + moff) * 128 + n] = (_Float16)v1;
    }
  }
}

// ---------------------------------------------------------------------------
// Gather 32 rows -> 3-layer 128x128 MLP (WMMA) -> max over the 32 samples.
// One group per block, 128 threads (4 waves).
// ---------------------------------------------------------------------------
__global__ __launch_bounds__(128)
void group_mlp_max_wmma(const float* __restrict__ fea, const int* __restrict__ gidx,
                        const _Float16* __restrict__ Wf,
                        const float* __restrict__ b0, const float* __restrict__ b1,
                        const float* __restrict__ b2, float* __restrict__ out) {
  __shared__ alignas(64) _Float16 Xa[32 * 128];
  __shared__ alignas(64) _Float16 Xb[32 * 128];
  int g = blockIdx.x;
  int lane = threadIdx.x & 31, wv = threadIdx.x >> 5;
  {
    int r = threadIdx.x >> 2;
    int srow = gidx[g * 32 + r];
    int c0 = (threadIdx.x & 3) * 32;
    const float* p = fea + (size_t)srow * 128 + c0;
    for (int j = 0; j < 32; j++) Xa[r * 128 + c0 + j] = (_Float16)p[j];
  }
  __syncthreads();
  mlp_layer128(Xa, Wf,          b0, Xb, lane, wv);
  __syncthreads();
  mlp_layer128(Xb, Wf + 16384,  b1, Xa, lane, wv);
  __syncthreads();
  // final layer fused with max over 32 rows
  for (int nt = wv; nt < 8; nt += 4) {
    v8f c0 = {0.f,0.f,0.f,0.f,0.f,0.f,0.f,0.f};
    v8f c1 = {0.f,0.f,0.f,0.f,0.f,0.f,0.f,0.f};
    for (int kt = 0; kt < 4; kt++) {
      v16h a0 = load_a_frag(Xa, 128, 0, kt, lane);
      v16h a1 = load_a_frag(Xa, 128, 1, kt, lane);
      v16h b  = *(const v16h*)(Wf + 32768 + (size_t)(kt * 8 + nt) * 512 + lane * 16);
      c0 = wmma16(a0, b, c0);
      c1 = wmma16(a1, b, c1);
    }
    int n = nt * 16 + (lane & 15);
    float bv = b2[n];
    float mx = 0.f;  // relu outputs are >= 0
    for (int r = 0; r < 8; r++) {
      mx = fmaxf(mx, fmaxf(c0[r] + bv, 0.f));
      mx = fmaxf(mx, fmaxf(c1[r] + bv, 0.f));
    }
    mx = fmaxf(mx, __shfl_xor(mx, 16, 32));
    if (lane < 16) out[(size_t)g * 128 + n] = mx;
  }
}

// ---------------------------------------------------------------------------
// Generic WMMA GEMM: out = act( [A1|A2] (MxK) @ W (KxN) + bias ).
// A f32 global, W pre-swizzled f16 fragments, out f32. K<=256 mult of 32,
// N mult of 16. Block = 128 threads = 4 waves, 32 rows per block.
// ---------------------------------------------------------------------------
__global__ __launch_bounds__(128)
void gemm_wmma(const float* __restrict__ A1, int K1,
               const float* __restrict__ A2, int K2,
               const _Float16* __restrict__ Wf, const float* __restrict__ bias,
               float* __restrict__ out, int M, int N, int act) {
  __shared__ alignas(64) _Float16 Xs[32 * 256];
  int K = K1 + K2;
  int lane = threadIdx.x & 31, wv = threadIdx.x >> 5;
  int rowbase = blockIdx.x * 32;
  {
    int r = threadIdx.x >> 2;
    int row = rowbase + r;
    int c0 = (threadIdx.x & 3) * (K >> 2);
    for (int j = 0; j < (K >> 2); j++) {
      int c = c0 + j;
      float v = 0.f;
      if (row < M) v = (c < K1) ? A1[(size_t)row * K1 + c]
                                : A2[(size_t)row * K2 + (c - K1)];
      Xs[r * K + c] = (_Float16)v;
    }
  }
  __syncthreads();
  int NT = N >> 4, KT = K >> 5;
  for (int nt = wv; nt < NT; nt += 4) {
    v8f c0 = {0.f,0.f,0.f,0.f,0.f,0.f,0.f,0.f};
    v8f c1 = {0.f,0.f,0.f,0.f,0.f,0.f,0.f,0.f};
    for (int kt = 0; kt < KT; kt++) {
      v16h a0 = load_a_frag(Xs, K, 0, kt, lane);
      v16h a1 = load_a_frag(Xs, K, 1, kt, lane);
      v16h b  = *(const v16h*)(Wf + (size_t)(kt * NT + nt) * 512 + lane * 16);
      c0 = wmma16(a0, b, c0);
      c1 = wmma16(a1, b, c1);
    }
    int n = nt * 16 + (lane & 15);
    float bv = bias ? bias[n] : 0.f;
    int moff = (lane & 16) ? 8 : 0;
    for (int r = 0; r < 8; r++) {
      float v0 = c0[r] + bv;
      float v1 = c1[r] + bv;
      if (act == 1) { v0 = fmaxf(v0, 0.f); v1 = fmaxf(v1, 0.f); }
      int row0 = rowbase + r + moff;
      int row1 = rowbase + 16 + r + moff;
      if (row0 < M) out[(size_t)row0 * N + n] = v0;
      if (row1 < M) out[(size_t)row1 * N + n] = v1;
    }
  }
}

// ---------------------------------------------------------------------------
// Small scalar GEMM for N not multiple of 16 (N=38, N=19). out=act(A@W+b).
// ---------------------------------------------------------------------------
__global__ void gemm_small(const float* __restrict__ A, const float* __restrict__ W,
                           const float* __restrict__ bias, float* __restrict__ out,
                           int M, int K, int N, int act) {
  int t = blockIdx.x * blockDim.x + threadIdx.x;
  if (t >= M * N) return;
  int m = t / N, n = t % N;
  float acc = bias ? bias[n] : 0.f;
  const float* a = A + (size_t)m * K;
  for (int k = 0; k < K; k++) acc += a[k] * W[(size_t)k * N + n];
  if (act == 1) acc = fmaxf(acc, 0.f);
  out[t] = acc;
}

// ---------------------------------------------------------------------------
// GAT attention scores: ssrc[i,h]=<h[i,h,:],a_src[h,:]>, sdst likewise.
// ---------------------------------------------------------------------------
__global__ void gat_scores(const float* __restrict__ h, const float* __restrict__ asrc,
                           const float* __restrict__ adst, int SB, int H, int C,
                           float* __restrict__ ssrc, float* __restrict__ sdst) {
  int t = blockIdx.x * blockDim.x + threadIdx.x;
  if (t >= SB * H) return;
  int i = t / H, hh = t % H;
  const float* hp = h + (size_t)i * H * C + hh * C;
  float s1 = 0.f, s2 = 0.f;
  for (int c = 0; c < C; c++) { s1 += hp[c] * asrc[hh * C + c]; s2 += hp[c] * adst[hh * C + c]; }
  ssrc[t] = s1; sdst[t] = s2;
}

// ---------------------------------------------------------------------------
// GAT softmax-aggregate, one node per block. Deterministic: single-thread
// ordered edge scan, ordered accumulation (no atomics). act: 0 none, 2 elu.
// ---------------------------------------------------------------------------
__global__ __launch_bounds__(128)
void gat_aggregate(const float* __restrict__ h, const int* __restrict__ esrc,
                   const int* __restrict__ edst, int E,
                   const float* __restrict__ ssrc, const float* __restrict__ sdst,
                   int H, int C, float* __restrict__ out, int act) {
  __shared__ int lst[4096];
  __shared__ int cnt_s;
  __shared__ float emax_s[2], den_s[2];
  int node = blockIdx.x;
  if (threadIdx.x == 0) {
    int c = 0;
    for (int e = 0; e < E; e++)
      if (edst[e] == node && c < 4096) lst[c++] = e;
    cnt_s = c;
  }
  __syncthreads();
  int cnt = cnt_s;
  if ((int)threadIdx.x < H) {
    int hh = threadIdx.x;
    float sd = sdst[node * H + hh];
    float m = -1e30f;
    for (int j = 0; j < cnt; j++) {
      float x = ssrc[esrc[lst[j]] * H + hh] + sd;
      x = x > 0.f ? x : 0.2f * x;
      m = fmaxf(m, x);
    }
    float den = 0.f;
    for (int j = 0; j < cnt; j++) {
      float x = ssrc[esrc[lst[j]] * H + hh] + sd;
      x = x > 0.f ? x : 0.2f * x;
      den += __expf(x - m);
    }
    emax_s[hh] = m;
    den_s[hh] = den + 1e-16f;
  }
  __syncthreads();
  int HC = H * C;
  for (int c = threadIdx.x; c < HC; c += 128) {
    int hh = c / C;
    float m = emax_s[hh], den = den_s[hh];
    float sd = sdst[node * H + hh];
    float acc = 0.f;
    for (int j = 0; j < cnt; j++) {
      int e = lst[j];
      int s = esrc[e];
      float x = ssrc[s * H + hh] + sd;
      x = x > 0.f ? x : 0.2f * x;
      float alpha = __expf(x - m) / den;
      acc += alpha * h[(size_t)s * HC + c];
    }
    if (act == 2) acc = acc > 0.f ? acc : (__expf(acc) - 1.f);
    out[(size_t)node * HC + c] = acc;
  }
}

__global__ void mean_heads(const float* __restrict__ in, float* __restrict__ out,
                           int SB, int H, int C) {
  int t = blockIdx.x * blockDim.x + threadIdx.x;
  if (t >= SB * C) return;
  int i = t / C, c = t % C;
  float s = 0.f;
  for (int h = 0; h < H; h++) s += in[(size_t)i * H * C + h * C + c];
  out[t] = s / (float)H;
}

__global__ void transpose_kernel(const float* __restrict__ in, float* __restrict__ out,
                                 int M, int N) {
  int t = blockIdx.x * blockDim.x + threadIdx.x;
  if (t >= M * N) return;
  int m = t / N, n = t % N;
  out[(size_t)n * M + m] = in[t];
}

// ---------------------------------------------------------------------------
// Host driver
// ---------------------------------------------------------------------------
extern "C" void kernel_launch(void* const* d_in, const int* in_sizes, int n_in,
                              void* d_out, int out_size, void* d_ws, size_t ws_size,
                              hipStream_t stream) {
  // Top-level inputs (setup_inputs dict order), then params pytree-flattened
  // (dict keys sorted, lists in order).
  const float* points_fea = (const float*)d_in[0];
  const float* points     = (const float*)d_in[1];
  const float* nodes_fea  = (const float*)d_in[2];
  const float* nodes      = (const float*)d_in[3];
  const int*   ei         = (const int*)d_in[4];
  const float* gl_b[3]  = {(const float*)d_in[5],  (const float*)d_in[6],  (const float*)d_in[7]};
  const float* gl_w[3]  = {(const float*)d_in[8],  (const float*)d_in[9],  (const float*)d_in[10]};
  const float* gnn_ad[3]= {(const float*)d_in[11], (const float*)d_in[12], (const float*)d_in[13]};
  const float* gnn_as[3]= {(const float*)d_in[14], (const float*)d_in[15], (const float*)d_in[16]};
  const float* gnn_w[3] = {(const float*)d_in[17], (const float*)d_in[18], (const float*)d_in[19]};
  const float* nh_ad[3] = {(const float*)d_in[20], (const float*)d_in[21], (const float*)d_in[22]};
  const float* nh_as[3] = {(const float*)d_in[23], (const float*)d_in[24], (const float*)d_in[25]};
  const float* nh_w[3]  = {(const float*)d_in[26], (const float*)d_in[27], (const float*)d_in[28]};
  const float* ph_b[3]  = {(const float*)d_in[29], (const float*)d_in[30], (const float*)d_in[31]};
  const float* ph_w[3]  = {(const float*)d_in[32], (const float*)d_in[33], (const float*)d_in[34]};
  const float* pm_b[3]  = {(const float*)d_in[35], (const float*)d_in[36], (const float*)d_in[37]};
  const float* pm_w[3]  = {(const float*)d_in[38], (const float*)d_in[39], (const float*)d_in[40]};

  const int E = in_sizes[4] / 2;
  const int* e_src = ei;
  const int* e_dst = ei + E;

  // workspace carve-up
  char* ws = (char*)d_ws;
  size_t off = 0;
  auto take = [&](size_t bytes) -> char* {
    char* p = ws + off;
    off = (off + bytes + 255) & ~(size_t)255;
    return p;
  };
  int* gi  = (int*)take((size_t)S_NODES * 32 * 4);
  int* gi2 = (int*)take((size_t)N_POINTS * 32 * 4);
  _Float16* glf   = (_Float16*)take(3 * 16384 * 2);
  _Float16* pmf0  = (_Float16*)take(32768 * 2);
  _Float16* pmf1  = (_Float16*)take(16384 * 2);
  _Float16* pmf2  = (_Float16*)take(16384 * 2);
  _Float16* gnnf0 = (_Float16*)take(32768 * 2);
  _Float16* gnnf1 = (_Float16*)take(16384 * 2);
  _Float16* gnnf2 = (_Float16*)take(16384 * 2);
  _Float16* nhf0  = (_Float16*)take(32768 * 2);
  _Float16* nhf1  = (_Float16*)take(32768 * 2);
  _Float16* phf0  = (_Float16*)take(16384 * 2);
  _Float16* phf1  = (_Float16*)take(8192 * 2);
  float* node_group = (float*)take((size_t)S_NODES * 128 * 4);
  float* gat_h   = (float*)take((size_t)S_NODES * 256 * 4);
  float* gscr    = (float*)take((size_t)S_NODES * 2 * 4);
  float* gdst    = (float*)take((size_t)S_NODES * 2 * 4);
  float* gat_x1  = (float*)take((size_t)S_NODES * 256 * 4);
  float* gat_x2  = (float*)take((size_t)S_NODES * 256 * 4);
  float* nh_tmp  = (float*)take((size_t)S_NODES * 38 * 4);
  float* pt_group= (float*)take((size_t)N_POINTS * 128 * 4);
  float* pmA     = (float*)take((size_t)N_POINTS * 128 * 4);
  float* pmB     = (float*)take((size_t)N_POINTS * 128 * 4);
  float* ptF     = (float*)take((size_t)N_POINTS * 128 * 4);
  float* h1      = (float*)take((size_t)N_POINTS * 128 * 4);
  float* h2      = (float*)take((size_t)N_POINTS * 64 * 4);
  float* h3      = (float*)take((size_t)N_POINTS * 19 * 4);

  // output sections (flat concat in return order)
  float* out_ptT  = (float*)d_out;                      // (128, 8192)
  float* out_nnf  = out_ptT + (size_t)128 * N_POINTS;   // (519, 128)
  float* out_pout = out_nnf + (size_t)S_NODES * 128;    // (19, 8192)
  float* out_nout = out_pout + (size_t)19 * N_POINTS;   // (519, 19)

  auto wfrag = [&](const float* W, _Float16* F, int K, int N) {
    int tot = K * N;
    wfrag_kernel<<<(tot + 255) / 256, 256, 0, stream>>>(W, F, K, N);
  };
  wfrag(gl_w[0], glf,           128, 128);
  wfrag(gl_w[1], glf + 16384,   128, 128);
  wfrag(gl_w[2], glf + 32768,   128, 128);
  wfrag(pm_w[0], pmf0,          256, 128);
  wfrag(pm_w[1], pmf1,          128, 128);
  wfrag(pm_w[2], pmf2,          128, 128);
  wfrag(gnn_w[0], gnnf0,        256, 128);
  wfrag(gnn_w[1], gnnf1,        128, 128);
  wfrag(gnn_w[2], gnnf2,        128, 128);
  wfrag(nh_w[0], nhf0,          128, 256);
  wfrag(nh_w[1], nhf1,          256, 128);
  wfrag(ph_w[0], phf0,          128, 128);
  wfrag(ph_w[1], phf1,          128, 64);

  // ball queries
  query_ball_kernel<<<(S_NODES + 127) / 128, 128, 0, stream>>>(points, N_POINTS, nodes, S_NODES, gi);
  query_ball_kernel<<<(N_POINTS + 127) / 128, 128, 0, stream>>>(nodes, S_NODES, points, N_POINTS, gi2);

  // grouped MLP+max (both share gl weights)
  group_mlp_max_wmma<<<S_NODES, 128, 0, stream>>>(points_fea, gi, glf, gl_b[0], gl_b[1], gl_b[2], node_group);
  group_mlp_max_wmma<<<N_POINTS, 128, 0, stream>>>(nodes_fea, gi2, glf, gl_b[0], gl_b[1], gl_b[2], pt_group);

  auto gemm = [&](const float* A1, int K1, const float* A2, int K2,
                  const _Float16* Wf, const float* bias, float* out, int M, int N, int act) {
    gemm_wmma<<<(M + 31) / 32, 128, 0, stream>>>(A1, K1, A2, K2, Wf, bias, out, M, N, act);
  };

  // ---- gnn GAT (H=1, C=128), input = [node_group | nodes_fea] ----
  gemm(node_group, 128, nodes_fea, 128, gnnf0, nullptr, gat_h, S_NODES, 128, 0);
  gat_scores<<<(S_NODES + 255) / 256, 256, 0, stream>>>(gat_h, gnn_as[0], gnn_ad[0], S_NODES, 1, 128, gscr, gdst);
  gat_aggregate<<<S_NODES, 128, 0, stream>>>(gat_h, e_src, e_dst, E, gscr, gdst, 1, 128, gat_x1, 2);
  gemm(gat_x1, 128, nullptr, 0, gnnf1, nullptr, gat_h, S_NODES, 128, 0);
  gat_scores<<<(S_NODES + 255) / 256, 256, 0, stream>>>(gat_h, gnn_as[1], gnn_ad[1], S_NODES, 1, 128, gscr, gdst);
  gat_aggregate<<<S_NODES, 128, 0, stream>>>(gat_h, e_src, e_dst, E, gscr, gdst, 1, 128, gat_x2, 2);
  gemm(gat_x2, 128, nullptr, 0, gnnf2, nullptr, gat_h, S_NODES, 128, 0);
  gat_scores<<<(S_NODES + 255) / 256, 256, 0, stream>>>(gat_h, gnn_as[2], gnn_ad[2], S_NODES, 1, 128, gscr, gdst);
  gat_aggregate<<<S_NODES, 128, 0, stream>>>(gat_h, e_src, e_dst, E, gscr, gdst, 1, 128, out_nnf, 0);

  // ---- point MLP: pt = relu-chain([pt_group | points_fea] @ pm) ----
  gemm(pt_group, 128, points_fea, 128, pmf0, pm_b[0], pmA, N_POINTS, 128, 1);
  gemm(pmA, 128, nullptr, 0, pmf1, pm_b[1], pmB, N_POINTS, 128, 1);
  gemm(pmB, 128, nullptr, 0, pmf2, pm_b[2], ptF, N_POINTS, 128, 1);
  transpose_kernel<<<((size_t)N_POINTS * 128 + 255) / 256, 256, 0, stream>>>(ptF, out_ptT, N_POINTS, 128);

  // ---- nh GAT on nodes_new_feature (H=2) ----
  gemm(out_nnf, 128, nullptr, 0, nhf0, nullptr, gat_h, S_NODES, 256, 0);
  gat_scores<<<(S_NODES * 2 + 255) / 256, 256, 0, stream>>>(gat_h, nh_as[0], nh_ad[0], S_NODES, 2, 128, gscr, gdst);
  gat_aggregate<<<S_NODES, 128, 0, stream>>>(gat_h, e_src, e_dst, E, gscr, gdst, 2, 128, gat_x1, 2);
  gemm(gat_x1, 256, nullptr, 0, nhf1, nullptr, gat_h, S_NODES, 128, 0);
  gat_scores<<<(S_NODES * 2 + 255) / 256, 256, 0, stream>>>(gat_h, nh_as[1], nh_ad[1], S_NODES, 2, 64, gscr, gdst);
  gat_aggregate<<<S_NODES, 128, 0, stream>>>(gat_h, e_src, e_dst, E, gscr, gdst, 2, 64, gat_x2, 2);
  gemm_small<<<((size_t)S_NODES * 38 + 255) / 256, 256, 0, stream>>>(gat_x2, nh_w[2], nullptr, nh_tmp, S_NODES, 128, 38, 0);
  gat_scores<<<(S_NODES * 2 + 255) / 256, 256, 0, stream>>>(nh_tmp, nh_as[2], nh_ad[2], S_NODES, 2, 19, gscr, gdst);
  gat_aggregate<<<S_NODES, 128, 0, stream>>>(nh_tmp, e_src, e_dst, E, gscr, gdst, 2, 19, gat_h, 0);
  mean_heads<<<((size_t)S_NODES * 19 + 255) / 256, 256, 0, stream>>>(gat_h, out_nout, S_NODES, 2, 19);

  // ---- ph head on pt ----
  gemm(ptF, 128, nullptr, 0, phf0, ph_b[0], h1, N_POINTS, 128, 1);
  gemm(h1, 128, nullptr, 0, phf1, ph_b[1], h2, N_POINTS, 64, 1);
  gemm_small<<<((size_t)N_POINTS * 19 + 255) / 256, 256, 0, stream>>>(h2, ph_w[2], ph_b[2], h3, N_POINTS, 64, 19, 0);
  transpose_kernel<<<((size_t)N_POINTS * 19 + 255) / 256, 256, 0, stream>>>(h3, out_pout, N_POINTS, 19);

  (void)n_in; (void)out_size; (void)ws_size;
}
